// BPNN_12678743458156
// MI455X (gfx1250) — matrix operand order repaired and checked by reading, not compiled, not run on hardware
//
#include <hip/hip_runtime.h>
#include <hip/hip_bf16.h>

typedef __attribute__((ext_vector_type(16))) _Float16 v16h;
typedef __attribute__((ext_vector_type(8)))  _Float16 v8h;
typedef __attribute__((ext_vector_type(4)))  _Float16 v4h;
typedef __attribute__((ext_vector_type(8)))  float    v8f;

#define N_ATOMS_C 50000
#define IN_DIM_C  128
#define HID_C     256
#define N_ELEM_C  4
#define N_IMG_C   500
#define NNZ_C     2000000

// ---------------------------------------------------------------------------
// Branch-free tanh: tanh(x) = 1 - 2/(exp2(2*log2e*x)+1). Saturates correctly
// at +-1 for large |x| (exp->inf or ->0) with no exec-mask slow path.
// ---------------------------------------------------------------------------
__device__ inline float fast_tanh(float x) {
    const float e = __builtin_amdgcn_exp2f(x * 2.885390081777927f);
    return 1.f - 2.f * __builtin_amdgcn_rcpf(e + 1.f);
}

// ---------------------------------------------------------------------------
// A-operand per CDNA5 ISA 7.12.2 "16-bit A-Matrix 16x32":
// lane row = lane&15, kb0 = (lane>=16)?8:0. The lane's 16 halves are two
// contiguous 8-half runs: K = kofs+kb0.. and K = kofs+16+kb0.. so the whole
// operand is two aligned ds_load_b128.
// ---------------------------------------------------------------------------
__device__ inline v16h loadA_fast(const _Float16* row_ptr, int kofs, int kb0) {
    v8h lo = *(const v8h*)(row_ptr + kofs + kb0);
    v8h hi = *(const v8h*)(row_ptr + kofs + 16 + kb0);
    return __builtin_shufflevector(lo, hi, 0, 1, 2, 3, 4, 5, 6, 7,
                                           8, 9, 10, 11, 12, 13, 14, 15);
}

// ---------------------------------------------------------------------------
// Pre-pack f32 weights -> f16 tiles in WMMA B-operand register order.
// B layout (32x16 f16): lane&15 = column, kb0 = (lane>>4)*8; lane holds
// K = kb0..kb0+7 then K = 16+kb0..16+kb0+7; lane operand contiguous (32B).
// dst tile index = blockIdx.x = (e*tilesN + t)*tilesK + kc.
// transposed=0: src is [E][K][N];  transposed=1: src is [E][N][K].
// ---------------------------------------------------------------------------
__global__ __launch_bounds__(32)
void bpnn_pack_kernel(const float* __restrict__ src, _Float16* __restrict__ dst,
                      int K, int N, int transposed) {
    const int lane   = threadIdx.x;
    const int tilesN = N / 16, tilesK = K / 32;
    const int b  = blockIdx.x;
    const int e  = b / (tilesN * tilesK);
    const int rm = b % (tilesN * tilesK);
    const int t  = rm / tilesK;
    const int kc = rm % tilesK;
    const int col   = t * 16 + (lane & 15);
    const int kbase = kc * 32 + ((lane >> 4) << 3);
    _Float16* d = dst + (size_t)b * 512 + lane * 16;
#pragma unroll
    for (int h = 0; h < 16; ++h) {
        const int k = kbase + h + ((h >> 3) << 3);
        const float v = transposed ? src[((size_t)e * N + col) * K + k]
                                   : src[((size_t)e * K + k) * N + col];
        d[h] = (_Float16)v;
    }
}

__global__ void bpnn_zero_kernel(float* __restrict__ p, int n) {
    const int i = blockIdx.x * blockDim.x + threadIdx.x;
    if (i < n) p[i] = 0.f;
}

// ---------------------------------------------------------------------------
// Fused MLP fwd + bwd: one wave32 per 16-atom tile.
//   energy += mask_e * MLP_e(fp)          (atomic into d_out[0..499])
//   g[n,:]  = sum_e d(mask_e*out_e)/dfp   (workspace, read by forces kernel)
// ---------------------------------------------------------------------------
__global__ __launch_bounds__(32)
void bpnn_mlp_kernel(const float* __restrict__ fp,
                     const int*  __restrict__ an,
                     const int*  __restrict__ img,
                     const float* __restrict__ b0,
                     const float* __restrict__ b1,
                     const float* __restrict__ W2,
                     const float* __restrict__ b2,
                     const _Float16* __restrict__ w0B,
                     const _Float16* __restrict__ w1B,
                     const _Float16* __restrict__ w1T,
                     const _Float16* __restrict__ w0T,
                     float* __restrict__ g,
                     float* __restrict__ energy) {
    __shared__ _Float16 fp_s[16][IN_DIM_C];   // fingerprints as f16
    __shared__ _Float16 h0_s[16][HID_C];      // tanh(z0)
    __shared__ _Float16 h1_s[16][HID_C];      // tanh(z1), reused as dz0
    __shared__ _Float16 dz_s[16][HID_C];      // dz1
    __shared__ float    g_s[16][IN_DIM_C];    // gradient accumulator
    __shared__ int      an_s[16];

    const int lane = threadIdx.x;
    const int a0   = blockIdx.x * 16;
    const int row  = lane & 15;         // A-layout row / C-layout column
    const int kb0  = ((lane >> 4) & 1) * 8;
    const int m    = (lane >> 4) * 8;   // C/D layout: row base for this half
    const int n    = lane & 15;         // C/D layout: column
    const int EZ[4] = {1, 6, 8, 29};

    // stage fingerprints tile as f16 (16x128): float4 in -> 4 halfs out
    {
        const float4* s = (const float4*)(fp + (size_t)a0 * IN_DIM_C);
        _Float16* d = &fp_s[0][0];
        for (int i = lane; i < 16 * IN_DIM_C / 4; i += 32) {
            const float4 v = s[i];
            v4h h;
            h[0] = (_Float16)v.x; h[1] = (_Float16)v.y;
            h[2] = (_Float16)v.z; h[3] = (_Float16)v.w;
            *(v4h*)(d + i * 4) = h;
        }
    }
    if (lane < 16) an_s[lane] = an[a0 + lane];
    {
        float4 z = {0.f, 0.f, 0.f, 0.f};
        float4* gz = (float4*)(&g_s[0][0]);
        for (int i = lane; i < 16 * IN_DIM_C / 4; i += 32) gz[i] = z;
    }
    __syncthreads();

    float oacc = 0.f;

    for (int e = 0; e < N_ELEM_C; ++e) {
        const float mrow = (an_s[row] == EZ[e]) ? 1.f : 0.f;   // mask for this lane's row

        // ---- layer 0: h0 = tanh(fp @ W0[e] + b0[e]) ----
        {
            v16h areg[IN_DIM_C / 32];
#pragma unroll
            for (int kc = 0; kc < IN_DIM_C / 32; ++kc)
                areg[kc] = loadA_fast(&fp_s[row][0], kc * 32, kb0);
            for (int t = 0; t < HID_C / 16; ++t) {
                const float bz = b0[e * HID_C + t * 16 + n];
                v8f acc;
#pragma unroll
                for (int r = 0; r < 8; ++r) acc[r] = bz;
                const v16h* bbase = (const v16h*)w0B +
                    ((size_t)(e * 16 + t) * (IN_DIM_C / 32)) * 32 + lane;
                v16h bcur = bbase[0];
#pragma unroll
                for (int kc = 0; kc < IN_DIM_C / 32; ++kc) {
                    const v16h bnxt = (kc + 1 < IN_DIM_C / 32) ? bbase[(kc + 1) * 32] : bcur;
                    acc = __builtin_amdgcn_wmma_f32_16x16x32_f16(false, areg[kc], false, bcur,
                                                                 (short)0, acc, false, false);
                    bcur = bnxt;
                }
#pragma unroll
                for (int r = 0; r < 8; ++r)
                    h0_s[m + r][t * 16 + n] = (_Float16)fast_tanh(acc[r]);
            }
        }
        __syncthreads();

        // ---- layer 1: h1 = tanh(h0 @ W1[e] + b1[e]) ----
        {
            v16h areg[HID_C / 32];
#pragma unroll
            for (int kc = 0; kc < HID_C / 32; ++kc)
                areg[kc] = loadA_fast(&h0_s[row][0], kc * 32, kb0);
            for (int t = 0; t < HID_C / 16; ++t) {
                const float bz = b1[e * HID_C + t * 16 + n];
                v8f acc;
#pragma unroll
                for (int r = 0; r < 8; ++r) acc[r] = bz;
                const v16h* bbase = (const v16h*)w1B +
                    ((size_t)(e * 16 + t) * (HID_C / 32)) * 32 + lane;
                v16h bcur = bbase[0];
#pragma unroll
                for (int kc = 0; kc < HID_C / 32; ++kc) {
                    const v16h bnxt = (kc + 1 < HID_C / 32) ? bbase[(kc + 1) * 32] : bcur;
                    acc = __builtin_amdgcn_wmma_f32_16x16x32_f16(false, areg[kc], false, bcur,
                                                                 (short)0, acc, false, false);
                    bcur = bnxt;
                }
#pragma unroll
                for (int r = 0; r < 8; ++r)
                    h1_s[m + r][t * 16 + n] = (_Float16)fast_tanh(acc[r]);
            }
        }
        __syncthreads();

        // ---- output: o = mask * (h1 . W2[e] + b2[e])  (8-wide dot) ----
        {
            float s = 0.f;
            const int k0 = (lane >> 4) * 128;
            const float* w2e = W2 + e * HID_C;
#pragma unroll
            for (int kv = 0; kv < 16; ++kv) {
                const v8h hv = *(const v8h*)(&h1_s[n][k0 + kv * 8]);
#pragma unroll
                for (int j = 0; j < 8; ++j)
                    s += (float)hv[j] * w2e[k0 + kv * 8 + j];
            }
            s += __shfl_xor(s, 16, 32);
            if (lane < 16) {
                const float mv = (an_s[n] == EZ[e]) ? 1.f : 0.f;
                oacc += mv * (s + b2[e]);
            }
        }

        // ---- dz1 = mask * W2[e] * (1 - h1^2)  (8-wide, flat split: row rn = iv/32)
        {
            const float* w2e = W2 + e * HID_C;
            // each row is 256/8 = 32 v8h, one per lane: flat split, one row per iteration
            for (int iv = lane; iv < 16 * HID_C / 8; iv += 32) {
                const int rn = iv >> 5;          // /32 v8h per row
                const int k  = (iv & 31) * 8;
                const float mv = (an_s[rn] == EZ[e]) ? 1.f : 0.f;
                const v8h hv = *(const v8h*)(&h1_s[rn][k]);
                v8h dv;
#pragma unroll
                for (int j = 0; j < 8; ++j) {
                    const float h1v = (float)hv[j];
                    dv[j] = (_Float16)(mv * w2e[k + j] * (1.f - h1v * h1v));
                }
                *(v8h*)(&dz_s[rn][k]) = dv;
            }
        }
        __syncthreads();

        // ---- dh0 = dz1 @ W1[e]^T ; dz0 = dh0 * (1 - h0^2) (into h1_s) ----
        {
            v16h areg[HID_C / 32];
#pragma unroll
            for (int kc = 0; kc < HID_C / 32; ++kc)
                areg[kc] = loadA_fast(&dz_s[row][0], kc * 32, kb0);
            for (int t = 0; t < HID_C / 16; ++t) {
                v8f acc;
#pragma unroll
                for (int r = 0; r < 8; ++r) acc[r] = 0.f;
                const v16h* bbase = (const v16h*)w1T +
                    ((size_t)(e * 16 + t) * (HID_C / 32)) * 32 + lane;
                v16h bcur = bbase[0];
#pragma unroll
                for (int kc = 0; kc < HID_C / 32; ++kc) {
                    const v16h bnxt = (kc + 1 < HID_C / 32) ? bbase[(kc + 1) * 32] : bcur;
                    acc = __builtin_amdgcn_wmma_f32_16x16x32_f16(false, areg[kc], false, bcur,
                                                                 (short)0, acc, false, false);
                    bcur = bnxt;
                }
#pragma unroll
                for (int r = 0; r < 8; ++r) {
                    const int col = t * 16 + n;
                    const float h0v = (float)h0_s[m + r][col];
                    h1_s[m + r][col] = (_Float16)(acc[r] * (1.f - h0v * h0v));
                }
            }
        }
        __syncthreads();

        // ---- g += dz0 @ W0[e]^T  (C operand carries the accumulator) ----
        {
            v16h areg[HID_C / 32];
#pragma unroll
            for (int kc = 0; kc < HID_C / 32; ++kc)
                areg[kc] = loadA_fast(&h1_s[row][0], kc * 32, kb0);   // dz0
            for (int t2 = 0; t2 < IN_DIM_C / 16; ++t2) {
                v8f acc;
#pragma unroll
                for (int r = 0; r < 8; ++r) acc[r] = g_s[m + r][t2 * 16 + n];
                const v16h* bbase = (const v16h*)w0T +
                    ((size_t)(e * (IN_DIM_C / 16) + t2) * (HID_C / 32)) * 32 + lane;
                v16h bcur = bbase[0];
#pragma unroll
                for (int kc = 0; kc < HID_C / 32; ++kc) {
                    const v16h bnxt = (kc + 1 < HID_C / 32) ? bbase[(kc + 1) * 32] : bcur;
                    acc = __builtin_amdgcn_wmma_f32_16x16x32_f16(false, areg[kc], false, bcur,
                                                                 (short)0, acc, false, false);
                    bcur = bnxt;
                }
#pragma unroll
                for (int r = 0; r < 8; ++r) g_s[m + r][t2 * 16 + n] = acc[r];
            }
        }
        __syncthreads();
    }

    // write gradient tile (float4)
    {
        const float4* gs = (const float4*)(&g_s[0][0]);
        float4* gd = (float4*)(g + (size_t)a0 * IN_DIM_C);
        for (int i = lane; i < 16 * IN_DIM_C / 4; i += 32) gd[i] = gs[i];
    }
    // per-image energy
    if (lane < 16) {
        const int atom = a0 + lane;
        atomicAdd(&energy[img[atom]], oacc);
    }
}

// ---------------------------------------------------------------------------
// forces[c] -= vals[i] * g[rows[i]]   (COO scatter, 2M nnz -> 150K slots)
// ---------------------------------------------------------------------------
__global__ void bpnn_forces_kernel(const int* __restrict__ rows,
                                   const int* __restrict__ cols,
                                   const float* __restrict__ vals,
                                   const float* __restrict__ g,
                                   float* __restrict__ forces, int nnz) {
    const int i = blockIdx.x * blockDim.x + threadIdx.x;
    if (i >= nnz) return;
    const int r = rows[i];
    const int c = cols[i];
    __builtin_prefetch(&g[r], 0, 0);
    atomicAdd(&forces[c], -vals[i] * g[r]);
}

extern "C" void kernel_launch(void* const* d_in, const int* in_sizes, int n_in,
                              void* d_out, int out_size, void* d_ws, size_t ws_size,
                              hipStream_t stream) {
    const float* fp   = (const float*)d_in[0];
    const int*   an   = (const int*)d_in[1];
    const int*   img  = (const int*)d_in[2];
    const int*   rows = (const int*)d_in[3];
    const int*   cols = (const int*)d_in[4];
    const float* vals = (const float*)d_in[5];
    const float* W0   = (const float*)d_in[6];
    const float* b0   = (const float*)d_in[7];
    const float* W1   = (const float*)d_in[8];
    const float* b1   = (const float*)d_in[9];
    const float* W2   = (const float*)d_in[10];
    const float* b2   = (const float*)d_in[11];

    float* out = (float*)d_out;        // [500] energy + [150000] forces
    char*  ws  = (char*)d_ws;

    // workspace layout (all offsets 256B aligned)
    const size_t g_bytes   = (size_t)N_ATOMS_C * IN_DIM_C * sizeof(float); // 25.6 MB
    const size_t w0B_bytes = (size_t)N_ELEM_C * IN_DIM_C * HID_C * 2;      // 256 KB
    const size_t w1B_bytes = (size_t)N_ELEM_C * HID_C * HID_C * 2;         // 512 KB
    float*    g   = (float*)ws;
    _Float16* w0B = (_Float16*)(ws + g_bytes);
    _Float16* w1B = (_Float16*)(ws + g_bytes + w0B_bytes);
    _Float16* w1T = (_Float16*)(ws + g_bytes + w0B_bytes + w1B_bytes);
    _Float16* w0T = (_Float16*)(ws + g_bytes + w0B_bytes + 2 * w1B_bytes);

    // 1) zero output (it is poisoned)
    bpnn_zero_kernel<<<(out_size + 255) / 256, 256, 0, stream>>>(out, out_size);

    // 2) pack weights into WMMA B-operand layout (f16)
    bpnn_pack_kernel<<<N_ELEM_C * (HID_C / 16) * (IN_DIM_C / 32), 32, 0, stream>>>(W0, w0B, IN_DIM_C, HID_C, 0);
    bpnn_pack_kernel<<<N_ELEM_C * (HID_C / 16) * (HID_C / 32),    32, 0, stream>>>(W1, w1B, HID_C,    HID_C, 0);
    bpnn_pack_kernel<<<N_ELEM_C * (HID_C / 16) * (HID_C / 32),    32, 0, stream>>>(W1, w1T, HID_C,    HID_C, 1);
    bpnn_pack_kernel<<<N_ELEM_C * (IN_DIM_C / 16) * (HID_C / 32), 32, 0, stream>>>(W0, w0T, HID_C, IN_DIM_C, 1);

    // 3) fused MLP fwd+bwd: energies into out[0..499], gradients into ws
    bpnn_mlp_kernel<<<N_ATOMS_C / 16, 32, 0, stream>>>(
        fp, an, img, b0, b1, W2, b2, w0B, w1B, w1T, w0T, g, out);

    // 4) sparse force scatter: out[500..150499]
    bpnn_forces_kernel<<<(NNZ_C + 255) / 256, 256, 0, stream>>>(
        rows, cols, vals, g, out + N_IMG_C, NNZ_C);
}
